// Block_9801115369805
// MI455X (gfx1250) — compile-verified, matrix operand order checked by reading, not run on hardware
//
#include <hip/hip_runtime.h>

typedef _Float16 f16;
typedef __attribute__((ext_vector_type(16))) _Float16 v16h;
typedef __attribute__((ext_vector_type(8)))  float    v8f;

#define NNODES 50000
#define NEDGES 800000
#define FIN    128
#define FOUT   128
#define MT     4            // M-tiles (of 16 edges) per block -> 64 edges/block

union V16H { v16h v; uint4 q[2]; };
union H8   { f16 h[8]; uint4 q; };

__device__ __forceinline__ uint4 pack8(float4 a, float4 b) {
  H8 u;
  u.h[0] = (f16)a.x; u.h[1] = (f16)a.y; u.h[2] = (f16)a.z; u.h[3] = (f16)a.w;
  u.h[4] = (f16)b.x; u.h[5] = (f16)b.y; u.h[6] = (f16)b.z; u.h[7] = (f16)b.w;
  return u.q;
}

// Load a 16x32 f16 A-fragment for one lane from an LDS row-major tile.
// lane<16 holds K=[kb..kb+7] then [kb+16..kb+23]; lane>=16 holds +8 offsets.
__device__ __forceinline__ v16h ld_a_frag(const f16* lds, int row_stride,
                                          int M, int hi, int kbase) {
  const int kb = kbase + hi * 8;
  V16H t;
  t.q[0] = *(const uint4*)(lds + M * row_stride + kb);
  t.q[1] = *(const uint4*)(lds + M * row_stride + kb + 16);
  return t.v;
}

// Load a pre-swizzled B fragment (32B contiguous per lane) from global.
__device__ __forceinline__ v16h ld_b_frag(const f16* __restrict__ wf,
                                          int frag, int lane) {
  const uint4* p = (const uint4*)(wf + ((size_t)(frag * 32 + lane)) * 16);
  V16H t;
  t.q[0] = p[0];
  t.q[1] = p[1];
  return t.v;
}

// Convert f32 row-major [K][N] weights to f16 in WMMA-B fragment order:
// out[((nt*ksteps + ks)*32 + lane)*16 + e]
__global__ __launch_bounds__(256) void pack_weights_kernel(
    const float* __restrict__ W, f16* __restrict__ out, int K, int N) {
  int o = blockIdx.x * blockDim.x + threadIdx.x;
  if (o >= K * N) return;
  int ksteps = K >> 5;
  int e    = o & 15;
  int lane = (o >> 4) & 31;
  int frag = o >> 9;
  int ks   = frag % ksteps;
  int nt   = frag / ksteps;
  int kl   = (e < 8 ? e : e + 8) + ((lane & 16) ? 8 : 0);
  int Kidx = ks * 32 + kl;
  int n    = nt * 16 + (lane & 15);
  out[o] = (f16)W[Kidx * N + n];
}

// Per-edge 2-layer MLP on 64-edge tiles (4 M-tiles per wave for B-fragment
// register reuse) + atomic mean-scatter.
__global__ __launch_bounds__(256) void edge_mlp_kernel(
    const float* __restrict__ x,
    const int* __restrict__ src,
    const int* __restrict__ dst,
    const f16* __restrict__ W1f,   // [nt=8][ks=8][lane=32][16]
    const float* __restrict__ b1,
    const f16* __restrict__ W2f,   // [nt=8][ks=4][lane=32][16]
    const float* __restrict__ b2,
    float* __restrict__ summed,
    float* __restrict__ cnt) {
  __shared__ f16 lds_e[MT * 16 * 256];   // [x_i | x_j - x_i] tiles, f16
  __shared__ f16 lds_h[MT * 16 * 128];   // hidden activations, f16
  __shared__ int lds_dst[MT * 16];

  const int tid   = threadIdx.x;
  const int lane  = tid & 31;
  const int wv    = tid >> 5;            // wave 0..7 -> N-tile
  const int ebase = blockIdx.x * (MT * 16);

  // ---- stage edge features (64 rows x 256 cols f16); 4 threads per row ----
  {
    const int r  = tid >> 2;             // edge row 0..63
    const int c0 = (tid & 3) * 32;       // 32 cols per thread
    const int e  = ebase + r;
    const int si = src[e];
    const int di = dst[e];
    if ((tid & 3) == 0) {
      lds_dst[r] = di;
      atomicAdd(&cnt[di], 1.0f);         // in-degree count (once per edge)
    }
    const float4* xi = (const float4*)(x + (size_t)di * FIN + c0);
    const float4* xj = (const float4*)(x + (size_t)si * FIN + c0);
    uint4* pe = (uint4*)(&lds_e[r * 256 + c0]);
    uint4* pd = (uint4*)(&lds_e[r * 256 + 128 + c0]);
#pragma unroll
    for (int k = 0; k < 4; ++k) {
      float4 a0 = xi[2 * k], a1 = xi[2 * k + 1];
      float4 j0 = xj[2 * k], j1 = xj[2 * k + 1];
      pe[k] = pack8(a0, a1);
      float4 d0 = make_float4(j0.x - a0.x, j0.y - a0.y, j0.z - a0.z, j0.w - a0.w);
      float4 d1 = make_float4(j1.x - a1.x, j1.y - a1.y, j1.z - a1.z, j1.w - a1.w);
      pd[k] = pack8(d0, d1);
    }
  }
  __syncthreads();

  const int M  = lane & 15;
  const int hi = (lane >> 4) & 1;
  const int n  = wv * 16 + (lane & 15);  // output column within 128

  // ---- GEMM1: [64x256] x [256x16-tile]; B-fragment reused across 4 M-tiles ----
  v8f acc[MT];
#pragma unroll
  for (int mt = 0; mt < MT; ++mt) acc[mt] = (v8f){};
#pragma unroll
  for (int ks = 0; ks < 8; ++ks) {
    v16h b = ld_b_frag(W1f, wv * 8 + ks, lane);
#pragma unroll
    for (int mt = 0; mt < MT; ++mt) {
      v16h a = ld_a_frag(lds_e + mt * 16 * 256, 256, M, hi, ks * 32);
      acc[mt] = __builtin_amdgcn_wmma_f32_16x16x32_f16(false, a, false, b,
                                                       (short)0, acc[mt],
                                                       false, false);
    }
  }
  {
    const float bias = b1[n];
#pragma unroll
    for (int mt = 0; mt < MT; ++mt) {
#pragma unroll
      for (int r = 0; r < 8; ++r) {
        float v = acc[mt][r] + bias;
        v = v > 0.0f ? v : 0.0f;                      // ReLU
        lds_h[(mt * 16 + r + 8 * hi) * 128 + n] = (f16)v;
      }
    }
  }
  __syncthreads();

  // ---- GEMM2: [64x128] x [128x16-tile] ----
  v8f acc2[MT];
#pragma unroll
  for (int mt = 0; mt < MT; ++mt) acc2[mt] = (v8f){};
#pragma unroll
  for (int ks = 0; ks < 4; ++ks) {
    v16h b = ld_b_frag(W2f, wv * 4 + ks, lane);
#pragma unroll
    for (int mt = 0; mt < MT; ++mt) {
      v16h a = ld_a_frag(lds_h + mt * 16 * 128, 128, M, hi, ks * 32);
      acc2[mt] = __builtin_amdgcn_wmma_f32_16x16x32_f16(false, a, false, b,
                                                        (short)0, acc2[mt],
                                                        false, false);
    }
  }
  {
    const float bias = b2[n];
#pragma unroll
    for (int mt = 0; mt < MT; ++mt) {
#pragma unroll
      for (int r = 0; r < 8; ++r) {
        const int row = mt * 16 + r + 8 * hi;
        atomicAdd(&summed[(size_t)lds_dst[row] * FOUT + n],
                  acc2[mt][r] + bias);
      }
    }
  }
}

// relu(mean) -> Linear(128,64) -> ReLU -> Linear(64,128) on 16-node tiles.
__global__ __launch_bounds__(256) void node_mlp_kernel(
    const float* __restrict__ summed,
    const float* __restrict__ cnt,
    const f16* __restrict__ W3f,   // [nt=4][ks=4][lane=32][16]
    const float* __restrict__ b3,
    const f16* __restrict__ W4f,   // [nt=8][ks=2][lane=32][16]
    const float* __restrict__ b4,
    float* __restrict__ out) {
  __shared__ f16 lds_a[16 * 128];
  __shared__ f16 lds_m[16 * 64];

  const int tid   = threadIdx.x;
  const int lane  = tid & 31;
  const int wv    = tid >> 5;
  const int nbase = blockIdx.x * 16;

  // ---- stage relu(mean) as f16 ----
  {
    const int r    = tid >> 4;
    const int c0   = (tid & 15) * 8;
    const int node = nbase + r;
    float cv  = cnt[node];
    float inv = 1.0f / (cv > 1.0f ? cv : 1.0f);
    const float4* sp = (const float4*)(summed + (size_t)node * FOUT + c0);
    float4 s0 = sp[0], s1 = sp[1];
    float4 m0 = make_float4(s0.x * inv, s0.y * inv, s0.z * inv, s0.w * inv);
    float4 m1 = make_float4(s1.x * inv, s1.y * inv, s1.z * inv, s1.w * inv);
    m0.x = m0.x > 0.f ? m0.x : 0.f; m0.y = m0.y > 0.f ? m0.y : 0.f;
    m0.z = m0.z > 0.f ? m0.z : 0.f; m0.w = m0.w > 0.f ? m0.w : 0.f;
    m1.x = m1.x > 0.f ? m1.x : 0.f; m1.y = m1.y > 0.f ? m1.y : 0.f;
    m1.z = m1.z > 0.f ? m1.z : 0.f; m1.w = m1.w > 0.f ? m1.w : 0.f;
    *(uint4*)(&lds_a[r * 128 + c0]) = pack8(m0, m1);
  }
  __syncthreads();

  const int M  = lane & 15;
  const int hi = (lane >> 4) & 1;

  // ---- GEMM3: [16x128] x [128x64], waves 0..3 (wave-uniform branch) ----
  if (wv < 4) {
    const int n = wv * 16 + (lane & 15);
    v8f acc = {};
#pragma unroll
    for (int ks = 0; ks < 4; ++ks) {
      v16h a = ld_a_frag(lds_a, 128, M, hi, ks * 32);
      v16h b = ld_b_frag(W3f, wv * 4 + ks, lane);
      acc = __builtin_amdgcn_wmma_f32_16x16x32_f16(false, a, false, b,
                                                   (short)0, acc, false, false);
    }
    const float bias = b3[n];
#pragma unroll
    for (int r = 0; r < 8; ++r) {
      float v = acc[r] + bias;
      v = v > 0.0f ? v : 0.0f;
      lds_m[(r + 8 * hi) * 64 + n] = (f16)v;
    }
  }
  __syncthreads();

  // ---- GEMM4: [16x64] x [64x128], all 8 waves ----
  {
    const int n = wv * 16 + (lane & 15);
    v8f acc = {};
#pragma unroll
    for (int ks = 0; ks < 2; ++ks) {
      v16h a = ld_a_frag(lds_m, 64, M, hi, ks * 32);
      v16h b = ld_b_frag(W4f, wv * 2 + ks, lane);
      acc = __builtin_amdgcn_wmma_f32_16x16x32_f16(false, a, false, b,
                                                   (short)0, acc, false, false);
    }
    const float bias = b4[n];
#pragma unroll
    for (int r = 0; r < 8; ++r) {
      out[(size_t)(nbase + r + 8 * hi) * FOUT + n] = acc[r] + bias;
    }
  }
}

extern "C" void kernel_launch(void* const* d_in, const int* in_sizes, int n_in,
                              void* d_out, int out_size, void* d_ws, size_t ws_size,
                              hipStream_t stream) {
  const float* x   = (const float*)d_in[0];
  const int*  eidx = (const int*)d_in[1];   // [2][NEDGES]: row0=src, row1=dst
  const float* W1  = (const float*)d_in[2];
  const float* b1  = (const float*)d_in[3];
  const float* W2  = (const float*)d_in[4];
  const float* b2  = (const float*)d_in[5];
  const float* W3  = (const float*)d_in[6];
  const float* b3  = (const float*)d_in[7];
  const float* W4  = (const float*)d_in[8];
  const float* b4  = (const float*)d_in[9];
  float* out = (float*)d_out;

  // Workspace layout (bytes):
  //   [0,            25,600,000) summed  f32 [NNODES*128]
  //   [25,600,000,   25,800,000) cnt     f32 [NNODES]
  //   [25,800,000, +65536)       W1f f16 (256x128)
  //   then W2f (32768 B), W3f (16384 B), W4f (16384 B)
  char* ws = (char*)d_ws;
  float* summed = (float*)ws;
  float* cnt    = (float*)(ws + 25600000);
  f16* W1f = (f16*)(ws + 25800000);
  f16* W2f = (f16*)(ws + 25800000 + 65536);
  f16* W3f = (f16*)(ws + 25800000 + 65536 + 32768);
  f16* W4f = (f16*)(ws + 25800000 + 65536 + 32768 + 16384);

  // Zero accumulators (summed + cnt are contiguous).
  hipMemsetAsync(summed, 0, (size_t)(NNODES * FOUT + NNODES) * sizeof(float), stream);

  pack_weights_kernel<<<(256 * 128 + 255) / 256, 256, 0, stream>>>(W1, W1f, 256, 128);
  pack_weights_kernel<<<(128 * 128 + 255) / 256, 256, 0, stream>>>(W2, W2f, 128, 128);
  pack_weights_kernel<<<(128 * 64  + 255) / 256, 256, 0, stream>>>(W3, W3f, 128, 64);
  pack_weights_kernel<<<(64  * 128 + 255) / 256, 256, 0, stream>>>(W4, W4f, 64, 128);

  edge_mlp_kernel<<<NEDGES / (MT * 16), 256, 0, stream>>>(
      x, eidx, eidx + NEDGES, W1f, b1, W2f, b2, summed, cnt);

  node_mlp_kernel<<<NNODES / 16, 256, 0, stream>>>(
      summed, cnt, W3f, b3, W4f, b4, out);
}
